// Net_65412351918223
// MI455X (gfx1250) — matrix operand order repaired and checked by reading, not compiled, not run on hardware
//
#include <hip/hip_runtime.h>
#include <hip/hip_bf16.h>

typedef __attribute__((ext_vector_type(16))) _Float16 v16h;
typedef __attribute__((ext_vector_type(8)))  _Float16 v8h;
typedef __attribute__((ext_vector_type(8)))  float    v8f;

#define NNODES 100000
#define NEDGES 600000

// guaranteed hardware f32 atomic add (GLOBAL_ATOMIC_ADD_F32, no return -> STOREcnt)
__device__ __forceinline__ void gadd_f32(float* p, float v) {
    asm volatile("global_atomic_add_f32 %0, %1, off" :: "v"(p), "v"(v) : "memory");
}

// ---------------- degree count ----------------
__global__ void deg_count(const long long* __restrict__ dst, float* __restrict__ deg, int E) {
    int t = blockIdx.x * blockDim.x + threadIdx.x;
    if (t < E) gadd_f32(&deg[(int)dst[t]], 1.0f);
}

// ---------------- scatter x (f32, D=9) ----------------
__global__ void scatter_x9(const long long* __restrict__ src, const long long* __restrict__ dst,
                           const float* __restrict__ x, float* __restrict__ agg, int E) {
    int t = blockIdx.x * blockDim.x + threadIdx.x;
    if (t >= E * 9) return;
    int e = t / 9, k = t - e * 9;
    gadd_f32(&agg[(int)dst[e] * 9 + k], x[(int)src[e] * 9 + k]);
}

// ---------------- scatter hidden (f16 source, D % 4 == 0) ----------------
__global__ void scatter_h(const long long* __restrict__ src, const long long* __restrict__ dst,
                          const _Float16* __restrict__ h, float* __restrict__ agg, int E, int D) {
    int chunks = D >> 2;
    int t = blockIdx.x * blockDim.x + threadIdx.x;
    if (t >= E * chunks) return;
    int e = t / chunks, c = t - e * chunks;
    int s = (int)src[e], d = (int)dst[e];
    const _Float16* hp = h + s * D + (c << 2);
    float* ap = agg + d * D + (c << 2);
    v8h hv4;  // 4 halfs via one 8-byte load
    *(uint64_t*)&hv4 = *(const uint64_t*)hp;
#pragma unroll
    for (int j = 0; j < 4; ++j) gadd_f32(ap + j, (float)hv4[j]);
}

// ---------------- pack weight [K,F] f32 -> WMMA B-fragment order f16 ----------------
// layout: out[((kt*NTF + tnn)*32 + lane)*16 + i] = W[(kt*32 + hlf*16 + i)*F + tnn*16 + (lane&15)]
__global__ void pack_b(const float* __restrict__ W, _Float16* __restrict__ out, int K, int F) {
    int t = blockIdx.x * blockDim.x + threadIdx.x;
    if (t >= K * F) return;
    int i    = t & 15;
    int lane = (t >> 4) & 31;
    int tile = t >> 9;
    int NTF = F >> 4;
    int tn = tile % NTF, kt = tile / NTF;
    int hlf = lane >> 4;
    int n = tn * 16 + (lane & 15);
    int k = kt * 32 + hlf * 16 + i;
    out[t] = (_Float16)W[k * F + n];
}

// ---------------- conv1: 9 -> 64, fused mean + dual linear + L2 norm + relu ----------------
__global__ void conv1_fused(const float* __restrict__ x, const float* __restrict__ agg,
                            const float* __restrict__ deg, const float* __restrict__ W1l,
                            const float* __restrict__ b1, const float* __restrict__ W1r,
                            _Float16* __restrict__ out) {
    int node = blockIdx.x;
    int f = threadIdx.x;                 // 0..63
    __shared__ float xa[9], aa[9], sq[64];
    if (f < 9) {
        float scale = 1.0f / fmaxf(deg[node], 1.0f);
        xa[f] = x[node * 9 + f];
        aa[f] = agg[node * 9 + f] * scale;
    }
    __syncthreads();
    float sum = b1[f];
#pragma unroll
    for (int k = 0; k < 9; ++k)
        sum += aa[k] * W1l[k * 64 + f] + xa[k] * W1r[k * 64 + f];
    sq[f] = sum * sum;
    __syncthreads();
    for (int s = 32; s > 0; s >>= 1) { if (f < s) sq[f] += sq[f + s]; __syncthreads(); }
    float inv = 1.0f / fmaxf(sqrtf(sq[0]), 1e-12f);
    out[node * 64 + f] = (_Float16)fmaxf(sum * inv, 0.0f);
}

// ---------------- WMMA GEMM: out = relu?( Aagg/deg @ Bl + Aself @ Br + bias ) ----------------
// Each wave computes MT row-tiles x NT col-tiles. B fragments are reused by MT WMMAs.
// HASAGG uses MT=1 (register pressure: f32 agg temps + converted frags), MLP uses MT=2.
template <int F, int NT, int MT, bool HASAGG>
__global__ void sage_gemm_wmma(const float* __restrict__ Aagg, const float* __restrict__ deg,
                               const _Float16* __restrict__ Blp,
                               const _Float16* __restrict__ Aself, const _Float16* __restrict__ Brp,
                               const float* __restrict__ bias,
                               _Float16* __restrict__ outh, float* __restrict__ outf,
                               int N, int K, int relu) {
    constexpr int NTF = F / 16;          // total column tiles in packed weights
    int lane = threadIdx.x & 31;
    int w = blockIdx.x * (blockDim.x >> 5) + (threadIdx.x >> 5);
    int tiles_n = NTF / NT;              // column groups
    int total = (N / (16 * MT)) * tiles_n;
    if (w >= total) return;
    int tm = w / tiles_n, tn = w - tm * tiles_n;
    int hlf = lane >> 4;                 // lane half: 0 or 1
    int nlo = lane & 15;

    int mrow[MT];
    float scale[MT];
#pragma unroll
    for (int i = 0; i < MT; ++i) {
        mrow[i] = tm * 16 * MT + i * 16 + (lane & 15);
        scale[i] = HASAGG ? 1.0f / fmaxf(deg[mrow[i]], 1.0f) : 0.0f;
    }

    v8f acc[MT][NT];
#pragma unroll
    for (int i = 0; i < MT; ++i)
#pragma unroll
        for (int t = 0; t < NT; ++t) acc[i][t] = (v8f){};

    const v16h* bl = (const v16h*)Blp;
    const v16h* br = (const v16h*)Brp;

    for (int kk = 0, kt = 0; kk < K; kk += 32, ++kt) {
        // A fragments: two contiguous 8-half runs per lane per row-tile
        v16h aself[MT];
#pragma unroll
        for (int i = 0; i < MT; ++i) {
            v8h slo = *(const v8h*)(Aself + mrow[i] * K + kk + hlf * 16);
            v8h shi = *(const v8h*)(Aself + mrow[i] * K + kk + hlf * 16 + 8);
            aself[i] = __builtin_shufflevector(slo, shi, 0, 1, 2, 3, 4, 5, 6, 7,
                                               8, 9, 10, 11, 12, 13, 14, 15);
        }
        if (HASAGG) {
            v16h aagg[MT];
#pragma unroll
            for (int i = 0; i < MT; ++i) {
                v8f flo = *(const v8f*)(Aagg + mrow[i] * K + kk + hlf * 16);
                v8f fhi = *(const v8f*)(Aagg + mrow[i] * K + kk + hlf * 16 + 8);
#pragma unroll
                for (int q = 0; q < 8; ++q) {
                    aagg[i][q]     = (_Float16)(flo[q] * scale[i]);
                    aagg[i][8 + q] = (_Float16)(fhi[q] * scale[i]);
                }
            }
#pragma unroll
            for (int t = 0; t < NT; ++t) {
                v16h b = bl[(kt * NTF + tn * NT + t) * 32 + lane];
#pragma unroll
                for (int i = 0; i < MT; ++i)
                    acc[i][t] = __builtin_amdgcn_wmma_f32_16x16x32_f16(
                        false, aagg[i], false, b, (short)0, acc[i][t], false, false);
            }
        }
#pragma unroll
        for (int t = 0; t < NT; ++t) {
            v16h b = br[(kt * NTF + tn * NT + t) * 32 + lane];
#pragma unroll
            for (int i = 0; i < MT; ++i)
                acc[i][t] = __builtin_amdgcn_wmma_f32_16x16x32_f16(
                    false, aself[i], false, b, (short)0, acc[i][t], false, false);
        }
    }
#pragma unroll
    for (int i = 0; i < MT; ++i)
#pragma unroll
        for (int t = 0; t < NT; ++t) {
            int n = tn * NT * 16 + t * 16 + nlo;
            float bv = bias[n];
#pragma unroll
            for (int j = 0; j < 8; ++j) {
                float v = acc[i][t][j] + bv;
                if (relu) v = fmaxf(v, 0.0f);
                int r = tm * 16 * MT + i * 16 + hlf * 8 + j;
                if (outh) outh[r * F + n] = (_Float16)v;
                if (outf) outf[r * F + n] = v;
            }
        }
}

// ---------------- head: 64 -> 3 + log_softmax ----------------
__global__ void head3(const _Float16* __restrict__ h, const float* __restrict__ W,
                      const float* __restrict__ b, float* __restrict__ out, int N) {
    int t = blockIdx.x * blockDim.x + threadIdx.x;
    if (t >= N) return;
    float a0 = b[0], a1 = b[1], a2 = b[2];
#pragma unroll 8
    for (int k = 0; k < 64; ++k) {
        float v = (float)h[t * 64 + k];
        a0 += v * W[k * 3 + 0];
        a1 += v * W[k * 3 + 1];
        a2 += v * W[k * 3 + 2];
    }
    float m = fmaxf(a0, fmaxf(a1, a2));
    float l = m + logf(expf(a0 - m) + expf(a1 - m) + expf(a2 - m));
    out[t * 3 + 0] = a0 - l;
    out[t * 3 + 1] = a1 - l;
    out[t * 3 + 2] = a2 - l;
}

extern "C" void kernel_launch(void* const* d_in, const int* in_sizes, int n_in,
                              void* d_out, int out_size, void* d_ws, size_t ws_size,
                              hipStream_t stream) {
    const float* x   = (const float*)d_in[0];
    const long long* ei = (const long long*)d_in[1];
    const long long* src = ei;
    const long long* dst = ei + NEDGES;
    const float* W1l = (const float*)d_in[2];
    const float* b1  = (const float*)d_in[3];
    const float* W1r = (const float*)d_in[4];
    const float* W2l = (const float*)d_in[5];
    const float* b2  = (const float*)d_in[6];
    const float* W2r = (const float*)d_in[7];
    const float* W3l = (const float*)d_in[8];
    const float* b3  = (const float*)d_in[9];
    const float* W3r = (const float*)d_in[10];
    const float* Wl1 = (const float*)d_in[11];
    const float* bl1 = (const float*)d_in[12];
    const float* Wl2 = (const float*)d_in[13];
    const float* bl2 = (const float*)d_in[14];
    const float* Wl3 = (const float*)d_in[15];
    const float* bl3 = (const float*)d_in[16];
    float* out = (float*)d_out;

    // workspace carving (~103 MB)
    char* p = (char*)d_ws;
    auto carve = [&](size_t bytes) { char* r = p; p += (bytes + 255) & ~(size_t)255; return (void*)r; };
    float*     deg  = (float*)    carve((size_t)NNODES * 4);
    float*     agg  = (float*)    carve((size_t)NNODES * 128 * 4);
    _Float16*  S1   = (_Float16*) carve((size_t)NNODES * 128 * 2);
    _Float16*  S2   = (_Float16*) carve((size_t)NNODES * 128 * 2);
    _Float16*  W2lh = (_Float16*) carve(64 * 128 * 2);
    _Float16*  W2rh = (_Float16*) carve(64 * 128 * 2);
    _Float16*  W3lh = (_Float16*) carve(128 * 128 * 2);
    _Float16*  W3rh = (_Float16*) carve(128 * 128 * 2);
    _Float16*  Wl1h = (_Float16*) carve(128 * 128 * 2);
    _Float16*  Wl2h = (_Float16*) carve(128 * 64 * 2);

    auto pack = [&](const float* W, _Float16* o, int K, int F) {
        int n = K * F;
        pack_b<<<(n + 255) / 256, 256, 0, stream>>>(W, o, K, F);
    };
    auto blocks_for = [&](int waves) { return (waves * 32 + 255) / 256; };
    int wv_agg = (NNODES / 16);      // HASAGG: MT=1, NT=8 -> one col group, wave per 16 rows
    int wv128  = (NNODES / 32) * 2;  // MLP F=128: MT=2, NT=4 -> 2 column groups
    int wv64   = (NNODES / 32) * 1;  // MLP F=64:  MT=2, NT=4 -> 1 column group

    // degrees
    hipMemsetAsync(deg, 0, (size_t)NNODES * 4, stream);
    deg_count<<<(NEDGES + 255) / 256, 256, 0, stream>>>(dst, deg, NEDGES);

    // weight pack (f32 -> f16 fragment order)
    pack(W2l, W2lh, 64, 128);  pack(W2r, W2rh, 64, 128);
    pack(W3l, W3lh, 128, 128); pack(W3r, W3rh, 128, 128);
    pack(Wl1, Wl1h, 128, 128); pack(Wl2, Wl2h, 128, 64);

    // conv1: 9 -> 64, L2-normalize + relu -> S1 (f16)
    hipMemsetAsync(agg, 0, (size_t)NNODES * 9 * 4, stream);
    scatter_x9<<<(NEDGES * 9 + 255) / 256, 256, 0, stream>>>(src, dst, x, agg, NEDGES);
    conv1_fused<<<NNODES, 64, 0, stream>>>(x, agg, deg, W1l, b1, W1r, S1);

    // conv2: 64 -> 128, relu -> S2
    hipMemsetAsync(agg, 0, (size_t)NNODES * 64 * 4, stream);
    scatter_h<<<(NEDGES * 16 + 255) / 256, 256, 0, stream>>>(src, dst, S1, agg, NEDGES, 64);
    sage_gemm_wmma<128, 8, 1, true><<<blocks_for(wv_agg), 256, 0, stream>>>(
        agg, deg, W2lh, S1, W2rh, b2, S2, nullptr, NNODES, 64, 1);

    // conv3: 128 -> 128, no relu -> S1
    hipMemsetAsync(agg, 0, (size_t)NNODES * 128 * 4, stream);
    scatter_h<<<(NEDGES * 32 + 255) / 256, 256, 0, stream>>>(src, dst, S2, agg, NEDGES, 128);
    sage_gemm_wmma<128, 8, 1, true><<<blocks_for(wv_agg), 256, 0, stream>>>(
        agg, deg, W3lh, S2, W3rh, b3, S1, nullptr, NNODES, 128, 0);

    // MLP head
    sage_gemm_wmma<128, 4, 2, false><<<blocks_for(wv128), 256, 0, stream>>>(
        nullptr, deg, nullptr, S1, Wl1h, bl1, S2, nullptr, NNODES, 128, 1);
    sage_gemm_wmma<64, 4, 2, false><<<blocks_for(wv64), 256, 0, stream>>>(
        nullptr, deg, nullptr, S2, Wl2h, bl2, S1, nullptr, NNODES, 128, 1);
    head3<<<(NNODES + 255) / 256, 256, 0, stream>>>(S1, Wl3, bl3, out, NNODES);
}
